// SarvamMoESparseMoeBlock_68410239091011
// MI455X (gfx1250) — compile-verified, matrix operand order checked by reading, not run on hardware
//
#include <hip/hip_runtime.h>
#include <math.h>

// ---------------------------------------------------------------------------
// Problem constants (match reference: T=1024, H=2048, I=1024, E=16, TOP_K=4)
// ---------------------------------------------------------------------------
#define T_TOK 1024
#define H_DIM 2048
#define I_DIM 1024
#define E_NUM 16
#define TOPK  4
#define NEXP  (E_NUM + 1)          // +1 = shared expert slot
#define HROWS_CAP 5376             // >= sum(roundup16(cnt)) + T for shared
#define MT    4                    // token tiles (16 rows each) per M-group
#define KBLK  256                  // K columns staged into LDS per block
#define ABUF_BYTES (64 * KBLK * 2) // one staged A block: 64 rows x KBLK bf16

typedef __bf16 bf16_t;
typedef bf16_t v16bf __attribute__((ext_vector_type(16)));
typedef bf16_t v8bf  __attribute__((ext_vector_type(8)));
typedef bf16_t v4bf  __attribute__((ext_vector_type(4)));
typedef bf16_t v2bf  __attribute__((ext_vector_type(2)));
typedef float  v8f   __attribute__((ext_vector_type(8)));
typedef float  v4f   __attribute__((ext_vector_type(4)));
typedef unsigned short v16us __attribute__((ext_vector_type(16)));

#if defined(__has_builtin)
#if __has_builtin(__builtin_amdgcn_cvt_pk_bf16_f32)
#define HAVE_PK_BF16 1
#endif
#endif

// fp32 -> bf16 (round-to-nearest-even), scalar (used in small epilogues)
__device__ __forceinline__ unsigned short f2bf(float f) {
    unsigned u = __builtin_bit_cast(unsigned, f);
    u += 0x7FFFu + ((u >> 16) & 1u);
    return (unsigned short)(u >> 16);
}

// A operand: 16-bit A-matrix 16x32 layout (ISA 7.12.2).
// lane l: row = l&15, kb = (l&16)?8:0 ; elements kb..kb+7 then kb+16..kb+23.
__device__ __forceinline__ v16bf load_a_bf16(const unsigned short* rowptr,
                                             int kk, int kb) {
    v8bf lo = *(const v8bf*)(const void*)(rowptr + kk + kb);
    v8bf hi = *(const v8bf*)(const void*)(rowptr + kk + kb + 16);
    return __builtin_shufflevector(lo, hi, 0, 1, 2, 3, 4, 5, 6, 7,
                                   8, 9, 10, 11, 12, 13, 14, 15);
}

// B operand: transposed weight row (column of B), fp32 in memory -> bf16.
__device__ __forceinline__ v16bf load_b_f32(const float* colptr, int kk, int kb) {
    v4f a = *(const v4f*)(const void*)(colptr + kk + kb);
    v4f b = *(const v4f*)(const void*)(colptr + kk + kb + 4);
    v4f c = *(const v4f*)(const void*)(colptr + kk + kb + 16);
    v4f d = *(const v4f*)(const void*)(colptr + kk + kb + 20);
#ifdef HAVE_PK_BF16
    v16bf r;
    v2bf p;
    p = __builtin_amdgcn_cvt_pk_bf16_f32(a[0], a[1]); r[0]  = p[0]; r[1]  = p[1];
    p = __builtin_amdgcn_cvt_pk_bf16_f32(a[2], a[3]); r[2]  = p[0]; r[3]  = p[1];
    p = __builtin_amdgcn_cvt_pk_bf16_f32(b[0], b[1]); r[4]  = p[0]; r[5]  = p[1];
    p = __builtin_amdgcn_cvt_pk_bf16_f32(b[2], b[3]); r[6]  = p[0]; r[7]  = p[1];
    p = __builtin_amdgcn_cvt_pk_bf16_f32(c[0], c[1]); r[8]  = p[0]; r[9]  = p[1];
    p = __builtin_amdgcn_cvt_pk_bf16_f32(c[2], c[3]); r[10] = p[0]; r[11] = p[1];
    p = __builtin_amdgcn_cvt_pk_bf16_f32(d[0], d[1]); r[12] = p[0]; r[13] = p[1];
    p = __builtin_amdgcn_cvt_pk_bf16_f32(d[2], d[3]); r[14] = p[0]; r[15] = p[1];
    return r;
#else
    // vector fptrunc -> should select v_cvt_pk_bf16_f32 on gfx1250
    v4bf pa = __builtin_convertvector(a, v4bf);
    v4bf pb = __builtin_convertvector(b, v4bf);
    v4bf pc = __builtin_convertvector(c, v4bf);
    v4bf pd = __builtin_convertvector(d, v4bf);
    v8bf lo = __builtin_shufflevector(pa, pb, 0, 1, 2, 3, 4, 5, 6, 7);
    v8bf hi = __builtin_shufflevector(pc, pd, 0, 1, 2, 3, 4, 5, 6, 7);
    return __builtin_shufflevector(lo, hi, 0, 1, 2, 3, 4, 5, 6, 7,
                                   8, 9, 10, 11, 12, 13, 14, 15);
#endif
}

__device__ __forceinline__ v8f wmma_bf16(v16bf a, v16bf b, v8f c) {
    return __builtin_amdgcn_wmma_f32_16x16x32_bf16(
        /*neg_a=*/false, a, /*neg_b=*/false, b,
        /*c_mod=*/(short)0, c, /*reuse_a=*/false, /*reuse_b=*/false);
}

// CDNA5 async global->LDS copy (16 bytes), tracked by ASYNCcnt.
__device__ __forceinline__ void async_copy_b128(unsigned lds_addr,
                                                const void* gptr) {
    asm volatile("global_load_async_to_lds_b128 %0, %1, off"
                 :: "v"(lds_addr), "v"(gptr)
                 : "memory");
}

__device__ __forceinline__ void wait_async0() {
    asm volatile("s_wait_asynccnt 0x0" ::: "memory");
}
// allow the newest 8 per-wave async ops to stay in flight (double buffering)
__device__ __forceinline__ void wait_async_le8() {
    asm volatile("s_wait_asynccnt 0x8" ::: "memory");
}

__device__ __forceinline__ float fast_silu_mul(float g, float u) {
    float sig = __builtin_amdgcn_rcpf(1.0f + __expf(-g));
    return g * sig * u;
}

// ---------------------------------------------------------------------------
// 0) zero output + bookkeeping arrays
// ---------------------------------------------------------------------------
__global__ void moe_zero_kernel(float* out, int* counts, int* cursors) {
    int i = blockIdx.x * blockDim.x + threadIdx.x;
    if (i < T_TOK * H_DIM) out[i] = 0.0f;
    if (i < NEXP) { counts[i] = 0; cursors[i] = 0; }
}

// ---------------------------------------------------------------------------
// 1) hidden_states fp32 -> bf16
// ---------------------------------------------------------------------------
__global__ void moe_cvt_x_kernel(const float* __restrict__ x,
                                 unsigned short* __restrict__ xb) {
    int i = blockIdx.x * blockDim.x + threadIdx.x;
    if (i < T_TOK * H_DIM) xb[i] = f2bf(x[i]);
}

// ---------------------------------------------------------------------------
// 2) router: logits -> sigmoid -> biased top-4 -> renormalize; count per expert
// ---------------------------------------------------------------------------
__global__ __launch_bounds__(256)
void moe_router_kernel(const float* __restrict__ x,
                       const float* __restrict__ gate_w,
                       const float* __restrict__ expert_bias,
                       int* __restrict__ topk_idx,
                       float* __restrict__ topk_w,
                       int* __restrict__ counts) {
    __shared__ float partial[E_NUM][17];
    const int t = blockIdx.x;
    const int tid = threadIdx.x;             // 256 threads
    const int e = tid & 15, lane = tid >> 4; // 16 threads per expert
    const float* xr = x + (size_t)t * H_DIM;
    const float* gr = gate_w + (size_t)e * H_DIM;
    float s = 0.0f;
    for (int i = lane; i < H_DIM; i += 16) s += xr[i] * gr[i];
    partial[e][lane] = s;
    __syncthreads();
    if (tid < E_NUM) {
        float sum = 0.0f;
        for (int i = 0; i < 16; ++i) sum += partial[tid][i];
        partial[tid][16] = 1.0f / (1.0f + __expf(-sum));  // sigmoid score
    }
    __syncthreads();
    if (tid == 0) {
        float sc[E_NUM], biased[E_NUM];
        for (int i = 0; i < E_NUM; ++i) {
            sc[i] = partial[i][16];
            biased[i] = sc[i] + expert_bias[i];
        }
        int   idx[TOPK];
        float wv[TOPK];
        float wsum = 0.0f;
        for (int k = 0; k < TOPK; ++k) {
            int best = -1; float bv = -1e30f;
            for (int i = 0; i < E_NUM; ++i)
                if (biased[i] > bv) { bv = biased[i]; best = i; }
            idx[k] = best; wv[k] = sc[best]; wsum += wv[k];
            biased[best] = -1e30f;
        }
        float inv = 1.0f / wsum;
        for (int k = 0; k < TOPK; ++k) {
            topk_idx[t * TOPK + k] = idx[k];
            topk_w[t * TOPK + k]   = wv[k] * inv;
            atomicAdd(&counts[idx[k]], 1);
        }
    }
}

// ---------------------------------------------------------------------------
// 3) 16-aligned exclusive prefix sum of counts -> offsets (slot 16 = shared)
// ---------------------------------------------------------------------------
__global__ void moe_offsets_kernel(const int* __restrict__ counts,
                                   int* __restrict__ offs) {
    if (blockIdx.x == 0 && threadIdx.x == 0) {
        int o = 0;
        for (int e = 0; e < E_NUM; ++e) {
            offs[e] = o;
            o += (counts[e] + 15) & ~15;
        }
        offs[E_NUM] = o;   // shared expert (count = T, already 16-aligned)
    }
}

// ---------------------------------------------------------------------------
// 4) scatter tokens into per-expert compact lists (+ identity list for shared)
// ---------------------------------------------------------------------------
__global__ void moe_scatter_kernel(const int* __restrict__ topk_idx,
                                   const float* __restrict__ topk_w,
                                   const int* __restrict__ offs,
                                   int* __restrict__ cursors,
                                   int* __restrict__ tok_id,
                                   float* __restrict__ tok_w) {
    int i = blockIdx.x * blockDim.x + threadIdx.x;
    if (i >= T_TOK * (TOPK + 1)) return;
    int t = i / (TOPK + 1), k = i % (TOPK + 1);
    if (k < TOPK) {
        int e = topk_idx[t * TOPK + k];
        int pos = atomicAdd(&cursors[e], 1);
        int o = offs[e] + pos;
        tok_id[o] = t;
        tok_w[o]  = topk_w[t * TOPK + k];
    } else {
        int o = offs[E_NUM] + t;
        tok_id[o] = t;
        tok_w[o]  = 1.0f;
    }
}

// ---------------------------------------------------------------------------
// 5) GEMM1: h = silu(x @ Wg^T) * (x @ Wu^T)  per expert, bf16 WMMA.
//    grid = NEXP * 8 workgroups, 256 threads (8 waves); each wave owns a
//    16-col tile of I.  M-groups of 64 gathered token rows; 64xKBLK A-blocks
//    double-buffered in LDS via global_load_async_to_lds_b128 so the next
//    block's HBM fetch overlaps WMMA on the current one.
// ---------------------------------------------------------------------------
__global__ __launch_bounds__(256)
void moe_gemm1_kernel(const unsigned short* __restrict__ xb,
                      const float* __restrict__ w_gate_up,
                      const float* __restrict__ shared_gate_up,
                      const int* __restrict__ tok_id,
                      const int* __restrict__ counts,
                      const int* __restrict__ offs,
                      unsigned short* __restrict__ h_buf) {
    __shared__ unsigned short sA[2][64 * KBLK];   // 2 x 32 KB staged A blocks
    __shared__ int sTok[64];

    const int e     = blockIdx.x >> 3;        // 0..16 (16 = shared)
    const int slice = blockIdx.x & 7;
    const int wave  = threadIdx.x >> 5;
    const int lane  = threadIdx.x & 31;
    const int lm    = lane & 15;
    const int kb    = (lane & 16) ? 8 : 0;
    const int mhi   = (lane & 16) ? 8 : 0;
    const int col   = slice * 128 + wave * 16 + lm;   // column of I

    const float* wbase = (e < E_NUM)
        ? (w_gate_up + (size_t)e * (2 * I_DIM) * H_DIM)
        : shared_gate_up;
    const int cnt = (e < E_NUM) ? counts[e] : T_TOK;
    const int off = offs[e];
    const int ntiles = (cnt + 15) >> 4;
    const int nmg = (ntiles + MT - 1) / MT;
    const int NKB = H_DIM / KBLK;             // 8 K-blocks

    const float* bg_ptr = wbase + (size_t)col * H_DIM;            // gate row
    const float* bu_ptr = wbase + (size_t)(I_DIM + col) * H_DIM;  // up row
    const unsigned lds_abase = (unsigned)(unsigned long long)(void*)&sA[0][0];

    for (int mg = 0; mg < nmg; ++mg) {
        const int row_base = mg * (MT * 16);
        if (threadIdx.x < 64) {
            int gidx = row_base + threadIdx.x;
            sTok[threadIdx.x] = (gidx < cnt) ? tok_id[off + gidx] : tok_id[off];
        }
        __syncthreads();

        // stage A block: 64 rows x KBLK bf16 = 2048 x 16B chunks, 8/thread
        auto stage = [&](int buf, int kc) {
#pragma unroll
            for (int t = 0; t < 8; ++t) {
                int ci  = threadIdx.x + t * 256;
                int row = ci >> 5;                 // 32 chunks per row
                int cch = ci & 31;
                const unsigned short* gp =
                    xb + (size_t)sTok[row] * H_DIM + kc + cch * 8;
                async_copy_b128(
                    lds_abase + (unsigned)(buf * ABUF_BYTES +
                                           row * (KBLK * 2) + cch * 16),
                    gp);
            }
        };

        v8f acc_g[MT], acc_u[MT];
#pragma unroll
        for (int mt = 0; mt < MT; ++mt) { acc_g[mt] = (v8f){}; acc_u[mt] = (v8f){}; }

        stage(0, 0);   // prologue
        for (int kbi = 0; kbi < NKB; ++kbi) {
            const int cur = kbi & 1;
            if (kbi + 1 < NKB) {
                stage((kbi + 1) & 1, (kbi + 1) * KBLK);
                wait_async_le8();   // current block done; next stays in flight
            } else {
                wait_async0();
            }
            __syncthreads();

            const int kc = kbi * KBLK;
            for (int k8 = 0; k8 < KBLK; k8 += 32) {
                v16bf bg = load_b_f32(bg_ptr, kc + k8, kb);
                v16bf bu = load_b_f32(bu_ptr, kc + k8, kb);
#pragma unroll
                for (int mt = 0; mt < MT; ++mt) {
                    const unsigned short* ap =
                        &sA[cur][(mt * 16 + lm) * KBLK + k8];
                    v16bf a = load_a_bf16(ap, 0, kb);
                    acc_g[mt] = wmma_bf16(a, bg, acc_g[mt]);
                    acc_u[mt] = wmma_bf16(a, bu, acc_u[mt]);
                }
            }
            __syncthreads();   // reads done before this buffer is re-staged
        }

#pragma unroll
        for (int mt = 0; mt < MT; ++mt) {
            int tile = mg * MT + mt;
            if (tile < ntiles) {
                int row0 = off + tile * 16 + mhi;
#pragma unroll
                for (int j = 0; j < 8; ++j) {
                    float hv = fast_silu_mul(acc_g[mt][j], acc_u[mt][j]);
                    h_buf[(size_t)(row0 + j) * I_DIM + col] = f2bf(hv);
                }
            }
        }
        __syncthreads();   // sTok reused next M-group
    }
}

// ---------------------------------------------------------------------------
// 6) GEMM2: out[t,:] += combine_w * (h @ Wd^T)   (atomic fp32 combine)
//    grid = NEXP * 16 workgroups, 256 threads; MT token tiles per pass;
//    contiguous h rows double-buffered in LDS via async copies.
// ---------------------------------------------------------------------------
__global__ __launch_bounds__(256)
void moe_gemm2_kernel(const unsigned short* __restrict__ h_buf,
                      const float* __restrict__ w_down,
                      const float* __restrict__ shared_down,
                      const int* __restrict__ tok_id,
                      const float* __restrict__ tok_w,
                      const int* __restrict__ counts,
                      const int* __restrict__ offs,
                      float* __restrict__ out) {
    __shared__ unsigned short sH[2][64 * KBLK];   // 2 x 32 KB staged h blocks

    const int e     = blockIdx.x >> 4;        // 0..16
    const int slice = blockIdx.x & 15;
    const int wave  = threadIdx.x >> 5;
    const int lane  = threadIdx.x & 31;
    const int lm    = lane & 15;
    const int kb    = (lane & 16) ? 8 : 0;
    const int mhi   = (lane & 16) ? 8 : 0;
    const int col   = slice * 128 + wave * 16 + lm;   // output column of H

    const float* wbase = (e < E_NUM)
        ? (w_down + (size_t)e * H_DIM * I_DIM)
        : shared_down;
    const int cnt = (e < E_NUM) ? counts[e] : T_TOK;
    const int off = offs[e];
    const int ntiles = (cnt + 15) >> 4;
    const int nmg = (ntiles + MT - 1) / MT;
    const int NKB = I_DIM / KBLK;             // 4 K-blocks
    const int hr_max = ntiles * 16 - 1;

    const float* b_ptr = wbase + (size_t)col * I_DIM;
    const unsigned lds_hbase = (unsigned)(unsigned long long)(void*)&sH[0][0];

    for (int mg = 0; mg < nmg; ++mg) {
        auto stage = [&](int buf, int kc) {
#pragma unroll
            for (int t = 0; t < 8; ++t) {
                int ci  = threadIdx.x + t * 256;
                int row = ci >> 5;
                int cch = ci & 31;
                int gr  = mg * 64 + row;
                if (gr > hr_max) gr = hr_max;
                const unsigned short* gp =
                    h_buf + (size_t)(off + gr) * I_DIM + kc + cch * 8;
                async_copy_b128(
                    lds_hbase + (unsigned)(buf * ABUF_BYTES +
                                           row * (KBLK * 2) + cch * 16),
                    gp);
            }
        };

        v8f acc[MT];
#pragma unroll
        for (int mt = 0; mt < MT; ++mt) acc[mt] = (v8f){};

        stage(0, 0);
        for (int kbi = 0; kbi < NKB; ++kbi) {
            const int cur = kbi & 1;
            if (kbi + 1 < NKB) {
                stage((kbi + 1) & 1, (kbi + 1) * KBLK);
                wait_async_le8();
            } else {
                wait_async0();
            }
            __syncthreads();

            const int kc = kbi * KBLK;
            for (int k8 = 0; k8 < KBLK; k8 += 32) {
                v16bf b = load_b_f32(b_ptr, kc + k8, kb);
#pragma unroll
                for (int mt = 0; mt < MT; ++mt) {
                    const unsigned short* ap =
                        &sH[cur][(mt * 16 + lm) * KBLK + k8];
                    v16bf a = load_a_bf16(ap, 0, kb);
                    acc[mt] = wmma_bf16(a, b, acc[mt]);
                }
            }
            __syncthreads();
        }

#pragma unroll
        for (int mt = 0; mt < MT; ++mt) {
            int tile = mg * MT + mt;
            if (tile >= ntiles) break;
#pragma unroll
            for (int j = 0; j < 8; ++j) {
                int gidx = tile * 16 + mhi + j;
                if (gidx < cnt) {
                    int   t = tok_id[off + gidx];
                    float w = tok_w[off + gidx];
                    atomicAdd(out + (size_t)t * H_DIM + col, w * acc[mt][j]);
                }
            }
        }
    }
}

// ---------------------------------------------------------------------------
// Launcher
// ---------------------------------------------------------------------------
extern "C" void kernel_launch(void* const* d_in, const int* in_sizes, int n_in,
                              void* d_out, int out_size, void* d_ws, size_t ws_size,
                              hipStream_t stream) {
    (void)in_sizes; (void)n_in; (void)out_size; (void)ws_size;
    const float* x              = (const float*)d_in[0];  // [T,H]
    const float* gate_w         = (const float*)d_in[1];  // [E,H]
    const float* expert_bias    = (const float*)d_in[2];  // [E]
    const float* w_gate_up      = (const float*)d_in[3];  // [E,2I,H]
    const float* w_down         = (const float*)d_in[4];  // [E,H,I]
    const float* shared_gate_up = (const float*)d_in[5];  // [2I,H]
    const float* shared_down    = (const float*)d_in[6];  // [H,I]
    float* out = (float*)d_out;                           // [T,H]

    // workspace layout
    char* ws = (char*)d_ws;
    size_t o = 0;
    unsigned short* xb    = (unsigned short*)(ws + o); o += (size_t)T_TOK * H_DIM * 2;     // 4 MB
    unsigned short* h_buf = (unsigned short*)(ws + o); o += (size_t)HROWS_CAP * I_DIM * 2; // 11 MB
    int*   topk_idx = (int*)  (ws + o); o += (size_t)T_TOK * TOPK * 4;
    float* topk_w   = (float*)(ws + o); o += (size_t)T_TOK * TOPK * 4;
    int*   tok_id   = (int*)  (ws + o); o += (size_t)HROWS_CAP * 4;
    float* tok_w    = (float*)(ws + o); o += (size_t)HROWS_CAP * 4;
    int*   counts   = (int*)  (ws + o); o += 128;
    int*   cursors  = (int*)  (ws + o); o += 128;
    int*   offs     = (int*)  (ws + o); o += 128;

    const int n_elems = T_TOK * H_DIM;

    moe_zero_kernel<<<(n_elems + 255) / 256, 256, 0, stream>>>(out, counts, cursors);
    moe_cvt_x_kernel<<<(n_elems + 255) / 256, 256, 0, stream>>>(x, xb);
    moe_router_kernel<<<T_TOK, 256, 0, stream>>>(x, gate_w, expert_bias,
                                                 topk_idx, topk_w, counts);
    moe_offsets_kernel<<<1, 32, 0, stream>>>(counts, offs);
    moe_scatter_kernel<<<(T_TOK * (TOPK + 1) + 255) / 256, 256, 0, stream>>>(
        topk_idx, topk_w, offs, cursors, tok_id, tok_w);
    moe_gemm1_kernel<<<NEXP * 8, 256, 0, stream>>>(
        xb, w_gate_up, shared_gate_up, tok_id, counts, offs, h_buf);
    moe_gemm2_kernel<<<NEXP * 16, 256, 0, stream>>>(
        h_buf, w_down, shared_down, tok_id, tok_w, counts, offs, out);
}
